// LSTM_13056700579951
// MI455X (gfx1250) — compile-verified, matrix operand order checked by reading, not run on hardware
//
#include <hip/hip_runtime.h>
#include <hip/hip_bf16.h>
#include <stdint.h>
#include <stddef.h>

// Problem constants (match the reference).
#define T_STEPS 1024
#define BATCH   64
#define IN_DIM  1024
#define HID     1024
#define G4H     4096   // 4*H

typedef __attribute__((ext_vector_type(16))) __bf16 v16bf;
typedef __attribute__((ext_vector_type(8)))  __bf16 v8bf;
typedef __attribute__((ext_vector_type(8)))  float  v8f;
typedef __attribute__((ext_vector_type(4)))  float  v4f;

static __device__ __forceinline__ float sigmoidf_(float x) {
  return 1.0f / (1.0f + __expf(-x));
}
static __device__ __forceinline__ float tanhf_(float x) {
  float e = __expf(-2.0f * x);
  return (1.0f - e) / (1.0f + e);
}

// Build a 16x32 (bf16) WMMA A/B fragment from a K-contiguous bf16 row.
// CDNA5 layout: lane L holds K = {kg*8..+7, 16+kg*8..+7}, kg = L>>4 —
// i.e. two contiguous 16-byte chunks per lane.
static __device__ __forceinline__ v16bf load_bf_frag(const __bf16* __restrict__ row,
                                                     int kA, int kB) {
  v8bf lo = *(const v8bf*)(row + kA);
  v8bf hi = *(const v8bf*)(row + kB);
  return __builtin_shufflevector(lo, hi, 0, 1, 2, 3, 4, 5, 6, 7,
                                 8, 9, 10, 11, 12, 13, 14, 15);
}

// Same, but from an fp32 row with on-the-fly bf16 conversion (for x).
static __device__ __forceinline__ v16bf load_f32_frag(const float* __restrict__ row,
                                                      int kA, int kB) {
  v4f a0 = *(const v4f*)(row + kA);
  v4f a1 = *(const v4f*)(row + kA + 4);
  v4f b0 = *(const v4f*)(row + kB);
  v4f b1 = *(const v4f*)(row + kB + 4);
  v16bf r;
#pragma unroll
  for (int j = 0; j < 4; ++j) {
    r[j]      = (__bf16)a0[j];
    r[j + 4]  = (__bf16)a1[j];
    r[j + 8]  = (__bf16)b0[j];
    r[j + 12] = (__bf16)b1[j];
  }
  return r;
}

// ---------------------------------------------------------------------------
// Kernel 0: one-time prep. Convert W_ih/W_hh to bf16, fuse biases, seed
// c-state and the h ping-pong buffers from h_0/c_0.
// ---------------------------------------------------------------------------
__global__ void lstm_init_kernel(const float* __restrict__ Wih,
                                 const float* __restrict__ Whh,
                                 const float* __restrict__ bih,
                                 const float* __restrict__ bhh,
                                 const float* __restrict__ h0,
                                 const float* __restrict__ c0,
                                 __bf16* __restrict__ wih_bf,
                                 __bf16* __restrict__ whh_bf,
                                 float*  __restrict__ bias,
                                 float*  __restrict__ cst,
                                 __bf16* __restrict__ hbf0,
                                 __bf16* __restrict__ hbf1) {
  const int stride = gridDim.x * blockDim.x;
  for (int i = blockIdx.x * blockDim.x + threadIdx.x; i < G4H * IN_DIM; i += stride) {
    wih_bf[i] = (__bf16)Wih[i];
    whh_bf[i] = (__bf16)Whh[i];   // G4H*HID == G4H*IN_DIM here
    if (i < G4H) bias[i] = bih[i] + bhh[i];
    if (i < BATCH * HID) {
      cst[i]  = c0[i];
      hbf0[i] = (__bf16)h0[i];
      hbf1[i] = (__bf16)0.0f;
    }
  }
}

// ---------------------------------------------------------------------------
// Kernel 1: time-parallel input projection.
//   pre[(t*B+b), n] = bias[n] + sum_k x[t,b,k] * W_ih[n,k]
// M = T*B = 65536, N = 4H = 4096, K = 1024.
// Block = 256 threads = 8 waves arranged 4(M) x 2(N); each wave owns a
// 16x128 output tile (8 accumulators). pre is written non-temporally (it is
// a 1 GiB stream consumed ~1024 launches later -> keep it out of L2 so
// W_ih/W_hh stay resident).
// ---------------------------------------------------------------------------
__global__ __launch_bounds__(256)
void lstm_pregate_gemm(const float*  __restrict__ x,
                       const __bf16* __restrict__ wih,
                       const float*  __restrict__ bias,
                       float*        __restrict__ pre) {
  const int lane = threadIdx.x & 31;
  const int wave = threadIdx.x >> 5;
  const int kg = lane >> 4, ln = lane & 15;
  const int mSub = wave & 3, nSub = wave >> 2;
  const int m0    = blockIdx.y * 64 + mSub * 16;      // [0, 65536)
  const int nBase = blockIdx.x * 256 + nSub * 128;    // [0, 4096)

  const float* xrow = x + (size_t)(m0 + ln) * IN_DIM;
  const __bf16* wrow[8];
#pragma unroll
  for (int nt = 0; nt < 8; ++nt)
    wrow[nt] = wih + (size_t)(nBase + nt * 16 + ln) * IN_DIM;

  const v8f vz = {0.f, 0.f, 0.f, 0.f, 0.f, 0.f, 0.f, 0.f};
  v8f acc[8];
#pragma unroll
  for (int nt = 0; nt < 8; ++nt) acc[nt] = vz;

  for (int k0 = 0; k0 < IN_DIM; k0 += 32) {
    const int kA = k0 + kg * 8;
    const int kB = k0 + 16 + kg * 8;
    const v16bf afrag = load_f32_frag(xrow, kA, kB);
#pragma unroll
    for (int nt = 0; nt < 8; ++nt) {
      const v16bf bfrag = load_bf_frag(wrow[nt], kA, kB);
      acc[nt] = __builtin_amdgcn_wmma_f32_16x16x32_bf16(
          false, afrag, false, bfrag, (short)0, acc[nt], false, false);
    }
  }

  // C/D layout: lane -> (row = v + 8*(lane>>4), col = lane&15).
#pragma unroll
  for (int nt = 0; nt < 8; ++nt) {
    const int gn = nBase + nt * 16 + ln;
    const float bs = bias[gn];
#pragma unroll
    for (int v = 0; v < 8; ++v) {
      const int gm = m0 + v + 8 * kg;
      __builtin_nontemporal_store(acc[nt][v] + bs, &pre[(size_t)gm * G4H + gn]);
    }
  }
}

// ---------------------------------------------------------------------------
// Kernel 2: one recurrent step (launched T times, in order, on the stream).
// gates[b, g*H+j] = pre[t,b,g*H+j] + sum_k h_{t-1}[b,k] * W_hh[g*H+j, k],
// then the fused LSTM cell update.
//
// Parallelism: grid = 256 blocks (4 b-tiles x 64 j-tiles), 4 waves/block
// doing a 4-way K-split (wave w covers K in [w*256, (w+1)*256)) -> 1024
// waves/step, LDS cross-wave reduction at the end.
//
// Latency hiding: the 4 W_hh B-fragments per K-step are double-buffered in
// LDS via GLOBAL_LOAD_ASYNC_TO_LDS_B128 (ASYNCcnt-tracked): prefetch the
// next K-step's fragments while the current buffer feeds the WMMAs.
// Async loads complete in order, so s_wait_asynccnt 0x8 (= the 8
// just-issued prefetches still allowed outstanding) guarantees the current
// buffer is resident.
// ---------------------------------------------------------------------------
__global__ __launch_bounds__(128)
void lstm_step(int t,
               const float*  __restrict__ pre,
               const __bf16* __restrict__ whh,
               const __bf16* __restrict__ hprev,
               __bf16*       __restrict__ hnext,
               float*        __restrict__ cst,
               float*        __restrict__ out) {
  // [buf][wave][gate][512 bf16 = 1 KiB]: async staging, 32 KiB.
  __shared__ __bf16 bstage[2][4][4][512] __attribute__((aligned(16)));
  // K-split partial sums from waves 1..3: 12 KiB.
  __shared__ float red[3][4][8][32];

  const int lane = threadIdx.x & 31;
  const int wave = threadIdx.x >> 5;     // K-split index
  const int kg = lane >> 4, ln = lane & 15;
  const int mt = blockIdx.x & 3;         // b-rows mt*16 .. +15
  const int j0 = (blockIdx.x >> 2) * 16; // h-cols j0 .. +15
  const int kBase = wave * (HID / 4);

  const __bf16* arow = hprev + (size_t)(mt * 16 + ln) * HID;
  const __bf16* brow[4];
#pragma unroll
  for (int g = 0; g < 4; ++g)
    brow[g] = whh + (size_t)(g * HID + j0 + ln) * HID;

  // Per-lane LDS byte offsets of this lane's chunk0 in each staging frag.
  uint32_t boff[2][4];
#pragma unroll
  for (int buf = 0; buf < 2; ++buf)
#pragma unroll
    for (int g = 0; g < 4; ++g)
      boff[buf][g] = (uint32_t)(uintptr_t)(&bstage[buf][wave][g][lane * 8]);

  const v8f vz = {0.f, 0.f, 0.f, 0.f, 0.f, 0.f, 0.f, 0.f};
  v8f acc[4];
#pragma unroll
  for (int g = 0; g < 4; ++g) acc[g] = vz;

  // ---- async prefetch of one K-step's 4 B-fragments into buffer `buf` ----
  auto prefetch = [&](int buf, int k0) {
    const int kA = k0 + kg * 8;
    const int kB = k0 + 16 + kg * 8;
#pragma unroll
    for (int g = 0; g < 4; ++g) {
      uint64_t p0 = (uint64_t)(uintptr_t)(brow[g] + kA);
      uint64_t p1 = (uint64_t)(uintptr_t)(brow[g] + kB);
      asm volatile("global_load_async_to_lds_b128 %0, %1, off"
                   :: "v"(boff[buf][g]), "v"(p0) : "memory");
      asm volatile("global_load_async_to_lds_b128 %0, %1, off"
                   :: "v"(boff[buf][g] + 512u), "v"(p1) : "memory");
    }
  };

  // ---- consume buffer `buf` with 4 WMMAs ----
  auto compute = [&](int buf, v16bf afrag) {
#pragma unroll
    for (int g = 0; g < 4; ++g) {
      v8bf lo = *(const v8bf*)(&bstage[buf][wave][g][lane * 8]);
      v8bf hi = *(const v8bf*)(&bstage[buf][wave][g][256 + lane * 8]);
      v16bf bfrag = __builtin_shufflevector(lo, hi, 0, 1, 2, 3, 4, 5, 6, 7,
                                            8, 9, 10, 11, 12, 13, 14, 15);
      acc[g] = __builtin_amdgcn_wmma_f32_16x16x32_bf16(
          false, afrag, false, bfrag, (short)0, acc[g], false, false);
    }
  };

  prefetch(0, kBase);
  v16bf a_cur = load_bf_frag(arow, kBase + kg * 8, kBase + 16 + kg * 8);

#pragma unroll
  for (int it = 0; it < 7; ++it) {
    const int kn = kBase + (it + 1) * 32;
    prefetch((it + 1) & 1, kn);
    v16bf a_next = load_bf_frag(arow, kn + kg * 8, kn + 16 + kg * 8);
    asm volatile("s_wait_asynccnt 0x8" ::: "memory");
    compute(it & 1, a_cur);
    a_cur = a_next;
  }
  asm volatile("s_wait_asynccnt 0x0" ::: "memory");
  compute(1, a_cur);

  // ---- cross-wave K reduction ----
  if (wave != 0) {
#pragma unroll
    for (int g = 0; g < 4; ++g)
#pragma unroll
      for (int v = 0; v < 8; ++v)
        red[wave - 1][g][v][lane] = acc[g][v];
  }
  __syncthreads();
  if (wave == 0) {
#pragma unroll
    for (int g = 0; g < 4; ++g)
#pragma unroll
      for (int v = 0; v < 8; ++v)
        acc[g][v] += red[0][g][v][lane] + red[1][g][v][lane] + red[2][g][v][lane];

    // ---- fused LSTM cell update (16x16 patch owned by wave 0) ----
#pragma unroll
    for (int v = 0; v < 8; ++v) {
      const int b   = mt * 16 + v + 8 * kg;
      const int col = j0 + ln;
      const float* pgr = pre + ((size_t)t * BATCH + b) * G4H;
      const float gi = acc[0][v] + __builtin_nontemporal_load(&pgr[col]);
      const float gf = acc[1][v] + __builtin_nontemporal_load(&pgr[HID + col]);
      const float gg = acc[2][v] + __builtin_nontemporal_load(&pgr[2 * HID + col]);
      const float go = acc[3][v] + __builtin_nontemporal_load(&pgr[3 * HID + col]);
      const float si = sigmoidf_(gi);
      const float sf = sigmoidf_(gf);
      const float sg = tanhf_(gg);
      const float so = sigmoidf_(go);
      const size_t ci = (size_t)b * HID + col;
      const float cn = sf * cst[ci] + si * sg;
      cst[ci] = cn;
      const float h = so * tanhf_(cn);
      __builtin_nontemporal_store(h, &out[((size_t)t * BATCH + b) * HID + col]);
      hnext[ci] = (__bf16)h;
    }
  }
}

// ---------------------------------------------------------------------------
// Kernel 3: emit (h_n, c_n) tail: h_n = output[T-1], c_n = c-state.
// ---------------------------------------------------------------------------
__global__ void lstm_tail(float* __restrict__ out, const float* __restrict__ cst) {
  const int i = blockIdx.x * blockDim.x + threadIdx.x;
  if (i < BATCH * HID) {
    const size_t TBH = (size_t)T_STEPS * BATCH * HID;
    out[TBH + i]               = out[TBH - (size_t)BATCH * HID + i];  // h_n
    out[TBH + BATCH * HID + i] = cst[i];                              // c_n
  }
}

// ---------------------------------------------------------------------------
// Launch. Inputs (setup_inputs order): x, h_0, c_0, W_ih, W_hh, b_ih, b_hh.
// Output: output[T,B,H] ++ h_n[1,B,H] ++ c_n[1,B,H], fp32.
// Workspace layout (needs ~1.02 GiB):
//   [0)            wih_bf : 4H*I  bf16  (8 MiB)
//   [8  MiB)       whh_bf : 4H*H  bf16  (8 MiB)
//   [16 MiB)       bias   : 4H    f32   (16 KiB)
//   [+16 KiB)      cstate : B*H   f32   (256 KiB)
//   [+256 KiB)     hbf0   : B*H   bf16  (128 KiB)
//   [+128 KiB)     hbf1   : B*H   bf16  (128 KiB)
//   [17317888)     pre    : T*B*4H f32  (1 GiB)
// ---------------------------------------------------------------------------
extern "C" void kernel_launch(void* const* d_in, const int* in_sizes, int n_in,
                              void* d_out, int out_size, void* d_ws, size_t ws_size,
                              hipStream_t stream) {
  (void)in_sizes; (void)n_in; (void)out_size; (void)ws_size;

  const float* x    = (const float*)d_in[0];
  const float* h0   = (const float*)d_in[1];
  const float* c0   = (const float*)d_in[2];
  const float* Wih  = (const float*)d_in[3];
  const float* Whh  = (const float*)d_in[4];
  const float* bih  = (const float*)d_in[5];
  const float* bhh  = (const float*)d_in[6];
  float* out = (float*)d_out;

  char* ws = (char*)d_ws;
  __bf16* wih_bf = (__bf16*)(ws);
  __bf16* whh_bf = (__bf16*)(ws + (size_t)8 * 1024 * 1024);
  float*  bias   = (float*) (ws + (size_t)16 * 1024 * 1024);
  float*  cst    = (float*) (ws + (size_t)16 * 1024 * 1024 + 16 * 1024);
  __bf16* hbf0   = (__bf16*)(ws + (size_t)16 * 1024 * 1024 + 16 * 1024 + 256 * 1024);
  __bf16* hbf1   = (__bf16*)(ws + (size_t)16 * 1024 * 1024 + 16 * 1024 + 256 * 1024 + 128 * 1024);
  float*  pre    = (float*) (ws + (size_t)17317888);

  // 0) prep: bf16 weights, fused bias, initial h/c state.
  lstm_init_kernel<<<2048, 256, 0, stream>>>(Wih, Whh, bih, bhh, h0, c0,
                                             wih_bf, whh_bf, bias, cst, hbf0, hbf1);

  // 1) time-parallel input projection (+bias) for all T at once.
  lstm_pregate_gemm<<<dim3(G4H / 256, (T_STEPS * BATCH) / 64), 256, 0, stream>>>(
      x, wih_bf, bias, pre);

  // 2) sequential recurrence, one fused kernel per step.
  for (int t = 0; t < T_STEPS; ++t) {
    __bf16* hp = (t & 1) ? hbf1 : hbf0;
    __bf16* hn = (t & 1) ? hbf0 : hbf1;
    lstm_step<<<dim3(256), 128, 0, stream>>>(t, pre, whh_bf, hp, hn, cst, out);
  }

  // 3) emit (h_n, c_n).
  lstm_tail<<<(BATCH * HID + 255) / 256, 256, 0, stream>>>(out, cst);
}
